// InputLayer_68212670595563
// MI455X (gfx1250) — matrix operand order repaired and checked by reading, not compiled, not run on hardware
//
#include <hip/hip_runtime.h>

// ---------------------------------------------------------------------------
// Sizes (match reference): B=64, T=512, D=512, H=1024, 4H=4096, K=D+H=1536
// d_out = [emb: B*T*D fp32][h_all: B*T*H fp32]
// ---------------------------------------------------------------------------
#define BB   64
#define TT   512
#define DD   512
#define HH   1024
#define GG   4096          // 4*H
#define KK   1536          // D + H
#define NBLK 64            // persistent grid size (one WG per hj tile)

// dynamic LDS: W resident (64 cols x 1536 k bf16 = 192 KiB) + X double buffer
#define LDS_W_ELEMS   ((size_t)64 * KK)          // 98304 bf16
#define LDS_X_ELEMS   ((size_t)2 * 64 * 64)      //  8192 bf16
#define LDS_BYTES     ((LDS_W_ELEMS + LDS_X_ELEMS) * 2)   // 212992 B <= 320 KiB

typedef __attribute__((ext_vector_type(16))) __bf16 v16bf;
typedef __attribute__((ext_vector_type(8)))  float  v8f;
typedef __attribute__((ext_vector_type(4)))  int    v4i;

union ABu { v16bf v; uint4 q[2]; };

static __device__ __forceinline__ unsigned short f2bf(float f) {
    unsigned u = __float_as_uint(f);
    u += 0x7FFFu + ((u >> 16) & 1u);          // round-to-nearest-even
    return (unsigned short)(u >> 16);
}
static __device__ __forceinline__ float sigf(float x) {
    return __builtin_amdgcn_rcpf(1.0f + __expf(-x));
}
static __device__ __forceinline__ float tanh_fast(float x) {
    float e = __expf(-2.0f * x);
    return (1.0f - e) * __builtin_amdgcn_rcpf(1.0f + e);
}

// ---- CDNA5 async global->LDS (ASYNCcnt) with graceful fallback -------------
#if defined(__has_builtin)
#  if __has_builtin(__builtin_amdgcn_global_load_async_to_lds_b128)
#    define HAVE_ASYNC_LDS 1
#  endif
#endif

static __device__ __forceinline__ void async_cp16(unsigned short* dst_lds,
                                                  const unsigned short* src_glob) {
#ifdef HAVE_ASYNC_LDS
    __builtin_amdgcn_global_load_async_to_lds_b128(
        (__attribute__((address_space(1))) v4i*)src_glob,
        (__attribute__((address_space(3))) v4i*)dst_lds, 0, 0);
#else
    *(uint4*)dst_lds = *(const uint4*)src_glob;
#endif
}
static __device__ __forceinline__ void async_wait0() {
#ifdef HAVE_ASYNC_LDS
#  if __has_builtin(__builtin_amdgcn_s_wait_asynccnt)
    __builtin_amdgcn_s_wait_asynccnt(0);
#  else
    asm volatile("s_wait_asynccnt 0x0" ::: "memory");
#  endif
#endif
}

// ---------------------------------------------------------------------------
// 0) init: zero grid-barrier counter, broadcast h0 into bf16 h state buffer
// ---------------------------------------------------------------------------
__global__ void init_state_k(unsigned* bar, unsigned short* hbuf, const float* h0) {
    unsigned idx = blockIdx.x * 256u + threadIdx.x;        // 65536 = B*H
    if (idx == 0) *bar = 0u;
    hbuf[idx] = f2bf(h0[idx & (HH - 1)]);
}

// ---------------------------------------------------------------------------
// 1) embedding gather: emb fp32 -> d_out, bf16 copy -> ws   (one row per WG)
// ---------------------------------------------------------------------------
__global__ void embed_k(const int* __restrict__ xb, const float* __restrict__ C,
                        float* __restrict__ emb_out, unsigned short* __restrict__ emb_bf) {
    const size_t row = blockIdx.x;                 // b*T + t
    const int    tid = threadIdx.x;                // 128 threads, 4 floats each
    const int    tok = xb[row];
    float4 v = ((const float4*)(C + (size_t)tok * DD))[tid];
    ((float4*)(emb_out + row * DD))[tid] = v;
    unsigned long long p =  (unsigned long long)f2bf(v.x)
                         | ((unsigned long long)f2bf(v.y) << 16)
                         | ((unsigned long long)f2bf(v.z) << 32)
                         | ((unsigned long long)f2bf(v.w) << 48);
    ((unsigned long long*)(emb_bf + row * DD))[tid] = p;
}

// ---------------------------------------------------------------------------
// 2) fused weight transpose: Wt[g][k] = (k<D ? U_w[k][g] : W_w[k-D][g]) as bf16
// ---------------------------------------------------------------------------
__global__ void wtrans_k(const float* __restrict__ U_w, const float* __restrict__ W_w,
                         unsigned short* __restrict__ Wt) {
    __shared__ float tile[32][33];
    const int kt = blockIdx.x * 32;                // k tile (48 blocks)
    const int gt = blockIdx.y * 32;                // g tile (128 blocks)
#pragma unroll
    for (int i = 0; i < 4; ++i) {
        int k = kt + threadIdx.y + i * 8;
        int g = gt + threadIdx.x;
        float v = (k < DD) ? U_w[(size_t)k * GG + g] : W_w[(size_t)(k - DD) * GG + g];
        tile[threadIdx.y + i * 8][threadIdx.x] = v;
    }
    __syncthreads();
#pragma unroll
    for (int i = 0; i < 4; ++i) {
        int g = gt + threadIdx.y + i * 8;
        int k = kt + threadIdx.x;
        Wt[(size_t)g * KK + k] = f2bf(tile[threadIdx.x][threadIdx.y + i * 8]);
    }
}

// ---------------------------------------------------------------------------
// helpers for the persistent kernel
// ---------------------------------------------------------------------------
static __device__ __forceinline__ void stage_x_chunk(
    unsigned short* XlBuf, const unsigned short* emb_bf, const unsigned short* hc,
    int t, int kc, int srow, int skof) {
    const unsigned short* sx = (kc < DD)
        ? emb_bf + ((size_t)srow * TT + t) * DD + kc + skof
        : hc + (size_t)srow * HH + (kc - DD) + skof;
    unsigned short* dx = XlBuf + srow * 64 + skof;
#pragma unroll
    for (int i = 0; i < 4; ++i) async_cp16(dx + i * 8, sx + i * 8);
}

// ---------------------------------------------------------------------------
// 3) persistent fused LSTM: 64 WGs x 128 threads (4 waves).
//    WG = hj (16 h-columns), wave = mi (16 batch rows).
//    W ([U;W] 64 gate-cols x 1536 K, bf16) lives in LDS for the whole kernel;
//    per step only the 8 KiB X chunk streams in via async global->LDS with
//    double buffering.  c state in accumulator registers; grid barrier / step.
// ---------------------------------------------------------------------------
__global__ void __launch_bounds__(128, 1)
lstm_persist_k(const unsigned short* __restrict__ emb_bf,
               const unsigned short* __restrict__ Wt,
               const float* __restrict__ U_b,
               const float* __restrict__ c0,
               unsigned short* __restrict__ h_b0,
               unsigned short* __restrict__ h_b1,
               float* __restrict__ h_out,
               unsigned* __restrict__ bar) {
    extern __shared__ __align__(16) unsigned short smem[];
    unsigned short* Wl = smem;                    // [64][KK]
    unsigned short* Xl = smem + LDS_W_ELEMS;      // [2][64][64]

    const int lane = threadIdx.x & 31;
    const int mi   = threadIdx.x >> 5;       // 0..3 : batch row tile
    const int lo   = lane & 15;
    const int hi   = lane >> 4;              // 0/1 : half-wave
    const int hj   = blockIdx.x;             // 0..63 : h column tile
    const int arow = mi * 16 + lo;           // A-operand LDS row

    // ---- one-time: pull this WG's 64 gate columns of Wt into LDS ----------
    {
        const int col  = threadIdx.x >> 1;                      // 0..63
        const int koff = (threadIdx.x & 1) * (KK / 2);          // 0 / 768
        const int gcol = ((col >> 4) << 10) + hj * 16 + (col & 15);
        const unsigned short* src = Wt + (size_t)gcol * KK + koff;
        unsigned short*       dst = Wl + (size_t)col * KK + koff;
        for (int i = 0; i < KK / 2; i += 8) async_cp16(dst + i, src + i);
        async_wait0();
    }
    __syncthreads();

    // hoisted per-lane constants
    float bias[4];
#pragma unroll
    for (int q = 0; q < 4; ++q) bias[q] = U_b[q * HH + hj * 16 + lo];
    const float c0v = c0[hj * 16 + lo];

    v8f creg;                                 // c state, accumulator layout
#pragma unroll
    for (int r = 0; r < 8; ++r) creg[r] = c0v;

    // staging roles (uniform per thread across steps)
    const int srow = threadIdx.x >> 1;                    // 0..63
    const int skof = (threadIdx.x & 1) * 32;              // 0 / 32

    for (int t = 0; t < TT; ++t) {
        const unsigned short* hc = (t & 1) ? h_b1 : h_b0;
        unsigned short*       hn = (t & 1) ? h_b0 : h_b1;

        v8f acc[4];
#pragma unroll
        for (int q = 0; q < 4; ++q)
#pragma unroll
            for (int r = 0; r < 8; ++r) acc[q][r] = bias[q];

        // prime X double buffer with chunk 0
        stage_x_chunk(Xl, emb_bf, hc, t, 0, srow, skof);
        async_wait0();
        __syncthreads();

        for (int kc = 0; kc < KK; kc += 64) {
            const int cur = (kc >> 6) & 1;
            unsigned short* Xc = Xl + (size_t)cur * 4096;
            // prefetch next chunk into the other buffer (async, overlaps WMMA)
            if (kc + 64 < KK)
                stage_x_chunk(Xl + (size_t)(cur ^ 1) * 4096, emb_bf, hc,
                              t, kc + 64, srow, skof);
#pragma unroll
            for (int kk = 0; kk < 64; kk += 32) {
                // A 16x32 bf16: lane holds row M=lo, K=[16*hi, 16*hi+16)
                ABu a;
                a.q[0] = *(const uint4*)&Xc[arow * 64 + kk + 16 * hi];
                a.q[1] = *(const uint4*)&Xc[arow * 64 + kk + 16 * hi + 8];
                // B 32x16 bf16: lane holds col N=lo, K=[16*hi, 16*hi+16)
                ABu b0, b1, b2, b3;
                const size_t wko = (size_t)kc + kk + 16 * hi;
                b0.q[0] = *(const uint4*)&Wl[(0 * 16 + lo) * (size_t)KK + wko];
                b0.q[1] = *(const uint4*)&Wl[(0 * 16 + lo) * (size_t)KK + wko + 8];
                b1.q[0] = *(const uint4*)&Wl[(1 * 16 + lo) * (size_t)KK + wko];
                b1.q[1] = *(const uint4*)&Wl[(1 * 16 + lo) * (size_t)KK + wko + 8];
                b2.q[0] = *(const uint4*)&Wl[(2 * 16 + lo) * (size_t)KK + wko];
                b2.q[1] = *(const uint4*)&Wl[(2 * 16 + lo) * (size_t)KK + wko + 8];
                b3.q[0] = *(const uint4*)&Wl[(3 * 16 + lo) * (size_t)KK + wko];
                b3.q[1] = *(const uint4*)&Wl[(3 * 16 + lo) * (size_t)KK + wko + 8];
                acc[0] = __builtin_amdgcn_wmma_f32_16x16x32_bf16(
                    false, a.v, false, b0.v, (short)0, acc[0], false, false);
                acc[1] = __builtin_amdgcn_wmma_f32_16x16x32_bf16(
                    false, a.v, false, b1.v, (short)0, acc[1], false, false);
                acc[2] = __builtin_amdgcn_wmma_f32_16x16x32_bf16(
                    false, a.v, false, b2.v, (short)0, acc[2], false, false);
                acc[3] = __builtin_amdgcn_wmma_f32_16x16x32_bf16(
                    false, a.v, false, b3.v, (short)0, acc[3], false, false);
            }
            async_wait0();          // staged chunk complete before buffer swap
            __syncthreads();
        }

        // gate nonlinearity + state update + store h (fp32 out, bf16 state)
#pragma unroll
        for (int r = 0; r < 8; ++r) {
            float iv = sigf(acc[0][r]);
            float fv = sigf(acc[1][r]);
            float gv = tanh_fast(acc[2][r]);
            float ov = sigf(acc[3][r]);
            float cc = fv * creg[r] + iv * gv;
            creg[r] = cc;
            float hh = ov * tanh_fast(cc);
            int brow = mi * 16 + r + 8 * hi;           // C/D layout row
            int col  = hj * 16 + lo;
            h_out[((size_t)brow * TT + t) * HH + col] = hh;
            hn[(size_t)brow * HH + col] = f2bf(hh);
        }

        // device-wide step barrier (monotonic counter, init'd per launch)
        __threadfence();
        __syncthreads();
        if (threadIdx.x == 0) {
            __hip_atomic_fetch_add(bar, 1u, __ATOMIC_ACQ_REL, __HIP_MEMORY_SCOPE_AGENT);
            const unsigned target = (unsigned)(t + 1) * NBLK;
            while (__hip_atomic_load(bar, __ATOMIC_ACQUIRE, __HIP_MEMORY_SCOPE_AGENT) < target)
                __builtin_amdgcn_s_sleep(1);
        }
        __syncthreads();
    }
}

// ---------------------------------------------------------------------------
extern "C" void kernel_launch(void* const* d_in, const int* in_sizes, int n_in,
                              void* d_out, int out_size, void* d_ws, size_t ws_size,
                              hipStream_t stream) {
    (void)in_sizes; (void)n_in; (void)out_size; (void)ws_size;
    const int*   xb  = (const int*)  d_in[0];
    const float* C   = (const float*)d_in[1];
    const float* U_w = (const float*)d_in[2];
    const float* U_b = (const float*)d_in[3];
    const float* W_w = (const float*)d_in[4];
    const float* h0  = (const float*)d_in[5];
    const float* c0  = (const float*)d_in[6];

    float* emb_out = (float*)d_out;
    float* h_out   = (float*)d_out + (size_t)BB * TT * DD;

    // workspace layout
    char* ws = (char*)d_ws;
    unsigned*       bar    = (unsigned*)ws;                               // 4 B (pad 256)
    unsigned short* emb_bf = (unsigned short*)(ws + 256);                 // B*T*D*2 = 32 MiB
    unsigned short* Wt     = emb_bf + (size_t)BB * TT * DD;               // 4096*1536*2 = 12 MiB
    unsigned short* h_b0   = Wt + (size_t)GG * KK;                        // 128 KiB
    unsigned short* h_b1   = h_b0 + (size_t)BB * HH;                      // 128 KiB

    init_state_k<<<(BB * HH) / 256, 256, 0, stream>>>(bar, h_b0, h0);
    embed_k<<<BB * TT, 128, 0, stream>>>(xb, C, emb_out, emb_bf);
    wtrans_k<<<dim3(KK / 32, GG / 32), dim3(32, 8), 0, stream>>>(U_w, W_w, Wt);
    lstm_persist_k<<<NBLK, 128, LDS_BYTES, stream>>>(emb_bf, Wt, U_b, c0,
                                                     h_b0, h_b1, h_out, bar);
}
